// ResonanceLayer_89034672046354
// MI455X (gfx1250) — compile-verified, matrix operand order checked by reading, not run on hardware
//
#include <hip/hip_runtime.h>
#include <hip/hip_bf16.h>
#include <math.h>

typedef __attribute__((ext_vector_type(8)))  _Float16 v8h;
typedef __attribute__((ext_vector_type(16))) _Float16 v16h;
typedef __attribute__((ext_vector_type(8)))  float    v8f;
typedef __attribute__((ext_vector_type(4)))  uint32_t u32x4;
typedef __attribute__((ext_vector_type(8)))  uint32_t u32x8;

#define N_SAMPLES   32768
#define WINDOW      512
#define STEP        256
#define N_FRAMES    128
#define N_COEFFS    257
#define N_RES       32
#define EXPR        4
#define CPD         16
#define CTRL_FRAMES 128
#define N_COMBO     (N_RES*EXPR)   /* 128 (r,e) pairs */
#define KDIM        544            /* 2*257=514 padded to 17*32 for WMMA K-steps */
#define PI_F        3.14159265358979323846f

/* ---- workspace layout (bytes, 256B aligned) ---- */
#define OFF_A       ((size_t)0)          /* f16 A[combo][t=128][k=544]   17,825,792 */
#define OFF_B       ((size_t)17825792)   /* f16 Bt[s=512][k=544]            557,056 */
#define OFF_FRAMES  ((size_t)18382848)   /* f32 frames[combo][128][512]  33,554,432 */
#define OFF_RES     ((size_t)51937280)   /* f32 res[combo][32768]        16,777,216 */
#define OFF_H       ((size_t)68714496)   /* f32 h[combo][32768]          16,777,216 */
#define OFF_NORM    ((size_t)85491712)   /* f32 norm[128] (padded)              512 */
#define OFF_VAL     ((size_t)85492224)   /* f32 val[8][32][128]             131,072 */
#define OFF_DSM     ((size_t)85623296)   /* f32 dsm[8][4][128]               16,384 */
/* total ~85.6 MB */

/* ------- async global->LDS copy of one f32 (ASYNCcnt path) ------- */
__device__ __forceinline__ void async_ld_f32(const float* gsrc, float* ldst) {
    uint32_t la = (uint32_t)(uintptr_t)ldst;      // low 32 bits = LDS byte address
    asm volatile("global_load_async_to_lds_b32 %0, %1, off"
                 :: "v"(la), "v"(gsrc) : "memory");
}
__device__ __forceinline__ void async_wait0() {
    asm volatile("s_wait_asynccnt 0x0" ::: "memory");
}

/* ---------------- init: zero accumulated output + norms ---------------- */
__global__ void k_init(float* __restrict__ out_sum, float* __restrict__ norm) {
    int i = blockIdx.x * blockDim.x + threadIdx.x;
    if (i < 8 * N_SAMPLES) out_sum[i] = 0.f;
    if (i < N_COMBO) norm[i] = 0.f;
}

/* ------- build A: decayed spectrum, interleaved (C,S), f16, K-padded ------- */
__global__ void k_buildA(const float* __restrict__ amp,
                         const float* __restrict__ phase,
                         const float* __restrict__ decay,
                         _Float16* __restrict__ A) {
    const int combo = blockIdx.x;              // r*4 + e
    const int r = combo >> 2, e = combo & 3;
    for (int kp = threadIdx.x; kp < KDIM / 2; kp += blockDim.x) {
        _Float16* dst = A + (size_t)combo * N_FRAMES * KDIM + 2 * kp;
        if (kp >= N_COEFFS) {   // zero K padding
            for (int t = 0; t < N_FRAMES; ++t) {
                dst[(size_t)t * KDIM]     = (_Float16)0.f;
                dst[(size_t)t * KDIM + 1] = (_Float16)0.f;
            }
            continue;
        }
        const size_t src = ((size_t)r * N_COEFFS + kp) * EXPR + e;
        float m  = fabsf(amp[src]) + 1e-12f;
        float ph = tanhf(phase[src]) * PI_F;
        float sg = 1.f / (1.f + expf(-decay[src]));
        float dd = 0.5f + sg * 0.45f;                       // BASE + sigmoid*0.5*0.9
        float w  = (kp == 0 || kp == N_COEFFS - 1) ? 1.f : 2.f;
        float scale = w * (1.f / (float)WINDOW);            // irfft 1/N and Hermitian x2
        float c = cosf(ph) * scale, s = sinf(ph) * scale;
        float mag = m;
        for (int t = 0; t < N_FRAMES; ++t) {
            mag *= dd;                                      // (m+1e-12)*d^(t+1)
            dst[(size_t)t * KDIM]     = (_Float16)(mag * c);
            dst[(size_t)t * KDIM + 1] = (_Float16)(mag * s);
        }
    }
}

/* ------- build B (shared basis), stored s-major, k-contiguous ------- */
__global__ void k_buildB(_Float16* __restrict__ B) {
    const int s = blockIdx.x;
    for (int kp = threadIdx.x; kp < KDIM / 2; kp += blockDim.x) {
        _Float16* dst = B + (size_t)s * KDIM + 2 * kp;
        if (kp >= N_COEFFS) { dst[0] = (_Float16)0.f; dst[1] = (_Float16)0.f; continue; }
        int mph = (kp * s) & (WINDOW - 1);                  // exact phase mod 512
        float th = (float)mph * (2.f * PI_F / (float)WINDOW);
        dst[0] = (_Float16)cosf(th);
        dst[1] = (_Float16)(-sinf(th));
    }
}

/* ------- WMMA fragment load: elems 0-7 from +0, 8-15 from +16 (ISA layout) ------- */
__device__ inline v16h load_frag(const _Float16* p) {
    v8h lo = *(const v8h*)(p);
    v8h hi = *(const v8h*)(p + 16);
    v16h r;
#pragma unroll
    for (int i = 0; i < 8; ++i) { r[i] = lo[i]; r[i + 8] = hi[i]; }
    return r;
}

/* ------- synthesis GEMM: frames[combo][t][s] = A[t,:] x B[:,s], f32 accum ------- */
__global__ void k_frames_wmma(const _Float16* __restrict__ A,
                              const _Float16* __restrict__ B,
                              float* __restrict__ frames) {
    const int wave  = (blockIdx.x * blockDim.x + threadIdx.x) >> 5;
    const int lane  = threadIdx.x & 31;
    const int combo = wave >> 8;
    const int rem   = wave & 255;
    const int tt = rem >> 5, ss = rem & 31;     // 8 t-tiles x 32 s-tiles of 16x16
    const int laneL = lane & 15, hi = lane >> 4;

    const _Float16* arow = A + (size_t)combo * N_FRAMES * KDIM
                             + (size_t)(tt * 16 + laneL) * KDIM + 8 * hi;
    const _Float16* brow = B + (size_t)(ss * 16 + laneL) * KDIM + 8 * hi;

    v8f acc = {};
    for (int k0 = 0; k0 < KDIM; k0 += 32) {
        v16h a = load_frag(arow + k0);
        v16h b = load_frag(brow + k0);
        acc = __builtin_amdgcn_wmma_f32_16x16x32_f16(false, a, false, b,
                                                     (short)0, acc, false, false);
    }
    float* out = frames + (size_t)combo * (N_FRAMES * WINDOW)
                        + (size_t)(tt * 16 + 8 * hi) * WINDOW + ss * 16 + laneL;
#pragma unroll
    for (int v = 0; v < 8; ++v) out[(size_t)v * WINDOW] = acc[v];
}

/* ------- overlap-add (hop 256) + sum-of-squares per combo ------- */
__global__ void k_oadd_norm(const float* __restrict__ frames,
                            float* __restrict__ res, float* __restrict__ norm) {
    const int combo = blockIdx.y;
    const int base  = blockIdx.x * 1024;
    const float* fr = frames + (size_t)combo * (N_FRAMES * WINDOW);
    float local = 0.f;
#pragma unroll
    for (int i = 0; i < 4; ++i) {
        int n = base + threadIdx.x + i * 256;
        int q = n >> 8, p = n & 255;
        float v = fr[(size_t)q * WINDOW + p];
        if (q >= 1) v += fr[(size_t)(q - 1) * WINDOW + STEP + p];
        res[(size_t)combo * N_SAMPLES + n] = v;
        local += v * v;
    }
    __shared__ float red[256];
    red[threadIdx.x] = local;
    __syncthreads();
    for (int off = 128; off > 0; off >>= 1) {
        if (threadIdx.x < off) red[threadIdx.x] += red[threadIdx.x + off];
        __syncthreads();
    }
    if (threadIdx.x == 0) atomicAdd(norm + combo, red[0]);
}

/* ------- h = (impulse FIR ⊛ res) / (||res|| + 1e-8), TDM-staged window ------- */
__global__ void k_build_h(const float* __restrict__ res,
                          const float* __restrict__ noise,
                          const float* __restrict__ norm,
                          float* __restrict__ h) {
    const int combo = blockIdx.y;
    const int base  = blockIdx.x * 1024;
    __shared__ float imp[128];
    __shared__ float buf[1152];
    if (threadIdx.x < 128) {
        int j = threadIdx.x;
        imp[j] = (0.54f - 0.46f * cosf(2.f * PI_F * (float)j / 128.f)) * noise[j];
    }
    /* Tensor Data Mover: wave 0 DMA-copies the res window into LDS (TENSORcnt). */
    const int gstart = (base == 0) ? 0 : (base - 127);
    const int nelem  = (base == 0) ? 1024 : 1151;
    const int ldsoff = (base == 0) ? 127 : 0;
    if (threadIdx.x < 32) {
        const float* gsrc = res + (size_t)combo * N_SAMPLES + gstart;
        uint64_t ga = (uint64_t)(uintptr_t)gsrc;
        uint32_t la = (uint32_t)(uintptr_t)(buf + ldsoff);
        uint32_t nd = (uint32_t)nelem;
        u32x4 g0;
        g0[0] = 1u;                                            // count=1 (valid), user mode
        g0[1] = la;                                            // lds_addr
        g0[2] = (uint32_t)ga;                                  // global_addr[31:0]
        g0[3] = (uint32_t)((ga >> 32) & 0x01FFFFFFu) | (2u << 30); // addr[56:32] | type=2
        u32x8 g1;
        g1[0] = 2u << 16;                                      // wg_mask=0, data_size=4B
        g1[1] = (nd & 0xFFFFu) << 16;                          // tensor_dim0[15:0]
        g1[2] = ((nd >> 16) & 0xFFFFu) | (1u << 16);           // dim0[31:16] | tensor_dim1=1
        g1[3] = nd << 16;                                      // tensor_dim1 hi=0 | tile_dim0
        g1[4] = 1u;                                            // tile_dim1=1, tile_dim2=0
        g1[5] = nd;                                            // tensor_dim0_stride[31:0]
        g1[6] = (nd & 0xFFFFu) << 16;                          // stride hi=0 | dim1_stride lo
        g1[7] = nd >> 16;                                      // dim1_stride rest
        asm volatile("tensor_load_to_lds %0, %1\n\t"
                     "s_wait_tensorcnt 0x0"
                     :: "s"(g0), "s"(g1) : "memory");
    }
    __syncthreads();
    if (base == 0) {                       // zero the pre-signal prefix (chunk 0 only)
        if (threadIdx.x < 127) buf[threadIdx.x] = 0.f;
        __syncthreads();
    }
    float inv = 1.f / (sqrtf(norm[combo]) + 1e-8f);
#pragma unroll
    for (int i = 0; i < 4; ++i) {
        int tl = threadIdx.x + i * 256;
        float acc = 0.f;
        for (int j = 0; j < 128; ++j) acc = fmaf(imp[j], buf[127 + tl - j], acc);
        h[(size_t)combo * N_SAMPLES + base + tl] = acc * inv;
    }
}

/* ------- router einsum; also writes the before_upsample output ------- */
__global__ void k_route(const float* __restrict__ control,
                        const float* __restrict__ router,
                        float* __restrict__ val,
                        float* __restrict__ out_before) {
    const int bn = blockIdx.x;
    const int f  = threadIdx.x;                 // 128
    __shared__ float ctrl[CPD * CTRL_FRAMES];
    __shared__ float rt[CPD * N_RES];
    for (int i = f; i < CPD * CTRL_FRAMES; i += 128)
        ctrl[i] = control[(size_t)bn * CPD * CTRL_FRAMES + i];
    for (int i = f; i < CPD * N_RES; i += 128) rt[i] = router[i];
    __syncthreads();
    for (int r = 0; r < N_RES; ++r) {
        float acc = 0.f;
#pragma unroll
        for (int c = 0; c < CPD; ++c) acc = fmaf(ctrl[c * CTRL_FRAMES + f], rt[c * N_RES + r], acc);
        size_t o = ((size_t)bn * N_RES + r) * CTRL_FRAMES + f;
        val[o] = acc;
        out_before[o] = acc;
    }
}

/* ------- deformation softmax over EXPR ------- */
__global__ void k_dsm(const float* __restrict__ def, float* __restrict__ dsm) {
    const int bn = blockIdx.x;
    const int f  = threadIdx.x;                 // 128
    float v[EXPR];
    float mx = -1e30f;
    for (int e = 0; e < EXPR; ++e) {
        v[e] = def[((size_t)bn * EXPR + e) * CTRL_FRAMES + f] + (e == 0 ? 1.f : 0.f);
        mx = fmaxf(mx, v[e]);
    }
    float s = 0.f;
    for (int e = 0; e < EXPR; ++e) { v[e] = expf(v[e] - mx); s += v[e]; }
    for (int e = 0; e < EXPR; ++e)
        dsm[((size_t)bn * EXPR + e) * CTRL_FRAMES + f] = v[e] / s;
}

/* ------- causal block conv + deformation mix + tanh + sum over r ------- */
__global__ void k_mix(const float* __restrict__ val,
                      const float* __restrict__ h,
                      const float* __restrict__ dsm,
                      const float* __restrict__ gains,
                      float* __restrict__ out_sum) {
    const int q0 = blockIdx.x * 16;             // q-tile
    const int r  = blockIdx.y;
    const int bn = blockIdx.z;
    const int p  = threadIdx.x;                 // 256 intra-hop phases
    __shared__ float vs[N_FRAMES];
    __shared__ float dl[EXPR * CTRL_FRAMES];
    /* async global->LDS staging of val row and deformation block (ASYNCcnt) */
    {
        const float* vsrc = val + ((size_t)bn * N_RES + r) * CTRL_FRAMES;
        const float* dsrc = dsm + (size_t)bn * EXPR * CTRL_FRAMES;
        if (threadIdx.x < 128) async_ld_f32(vsrc + threadIdx.x, vs + threadIdx.x);
        for (int i = threadIdx.x; i < EXPR * CTRL_FRAMES; i += 256)
            async_ld_f32(dsrc + i, dl + i);
        async_wait0();
    }
    __syncthreads();

    float g = fabsf(gains[r]);
    float x[16];
#pragma unroll
    for (int qi = 0; qi < 16; ++qi) x[qi] = 0.f;

    for (int e = 0; e < EXPR; ++e) {
        const float* hb = h + (size_t)(r * EXPR + e) * N_SAMPLES + p;  // L2-resident
        if (e + 1 < EXPR)   // warm WGP$ for next resonance row
            __builtin_prefetch(h + (size_t)(r * EXPR + e + 1) * N_SAMPLES + p, 0, 1);
        const float* de = dl + e * CTRL_FRAMES;
#pragma unroll
        for (int qi = 0; qi < 16; ++qi) {
            int q = q0 + qi;
            float acc = 0.f;
            for (int k = 0; k <= q; ++k)
                acc = fmaf(vs[k], hb[(size_t)(q - k) << 8], acc);
            float pos = ((float)(q * 256 + p) + 0.5f) * (1.f / 256.f) - 0.5f;
            pos = fminf(fmaxf(pos, 0.f), 127.f);
            int lo  = (int)pos;
            int hi2 = min(lo + 1, 127);
            float w = pos - (float)lo;
            float dv = de[lo] * (1.f - w) + de[hi2] * w;
            x[qi] = fmaf(dv, acc, x[qi]);
        }
    }
#pragma unroll
    for (int qi = 0; qi < 16; ++qi) {
        float y = tanhf(g * x[qi]);
        atomicAdd(out_sum + (size_t)bn * N_SAMPLES + (size_t)(q0 + qi) * 256 + p, y);
    }
}

extern "C" void kernel_launch(void* const* d_in, const int* in_sizes, int n_in,
                              void* d_out, int out_size, void* d_ws, size_t ws_size,
                              hipStream_t stream) {
    const float* control = (const float*)d_in[0];
    const float* def     = (const float*)d_in[1];
    const float* router  = (const float*)d_in[2];
    const float* gains   = (const float*)d_in[3];
    const float* amp     = (const float*)d_in[4];
    const float* phase   = (const float*)d_in[5];
    const float* decay   = (const float*)d_in[6];
    const float* noise   = (const float*)d_in[7];

    float* out = (float*)d_out;     // [0..262143]=summed, [262144..294911]=before_upsample
    char* ws = (char*)d_ws;
    _Float16* A   = (_Float16*)(ws + OFF_A);
    _Float16* B   = (_Float16*)(ws + OFF_B);
    float* frames = (float*)(ws + OFF_FRAMES);
    float* res    = (float*)(ws + OFF_RES);
    float* h      = (float*)(ws + OFF_H);
    float* norm   = (float*)(ws + OFF_NORM);
    float* val    = (float*)(ws + OFF_VAL);
    float* dsm    = (float*)(ws + OFF_DSM);

    k_init       <<<1024, 256, 0, stream>>>(out, norm);
    k_buildA     <<<N_COMBO, 128, 0, stream>>>(amp, phase, decay, A);
    k_buildB     <<<WINDOW, 256, 0, stream>>>(B);
    k_frames_wmma<<<4096, 256, 0, stream>>>(A, B, frames);   // 32768 waves, 16x16 tiles
    k_oadd_norm  <<<dim3(32, N_COMBO), 256, 0, stream>>>(frames, res, norm);
    k_build_h    <<<dim3(32, N_COMBO), 256, 0, stream>>>(res, noise, norm, h);
    k_route      <<<8, 128, 0, stream>>>(control, router, val, out + 8 * N_SAMPLES);
    k_dsm        <<<8, 128, 0, stream>>>(def, dsm);
    k_mix        <<<dim3(8, N_RES, 8), 256, 0, stream>>>(val, h, dsm, gains, out);
}
